// MultiHeadAttentionWithBias_11519102288238
// MI455X (gfx1250) — compile-verified, matrix operand order checked by reading, not run on hardware
//
#include <hip/hip_runtime.h>

// ---------------------------------------------------------------------------
// MHA with additive bias, MI455X (gfx1250, wave32, WMMA + async-to-LDS).
// Pipeline:
//   1) fp32 -> bf16 conversions (x, qkv_w, out_w)
//   2) GEMM  qkv[8192,3072] = x[8192,1024] @ qkv_w[3072,1024]^T + qkv_b  (bf16 out)
//      - global_load_async_to_lds_b128 double-buffered tiles
//   3) flash attention per (b,h): online softmax, bias added to scores
//      - async K tiles, global_prefetch on the 256MB bias stream
//   4) GEMM  out[8192,1024] = attn[8192,1024] @ out_w[1024,1024]^T + out_b (f32 out)
// All matmuls use v_wmma_f32_16x16x32_bf16, f32 accumulate.
// ---------------------------------------------------------------------------

typedef unsigned short u16;
typedef __attribute__((ext_vector_type(16))) __bf16 v16bf;
typedef __attribute__((ext_vector_type(8)))  float  v8f;

union FragBF { uint4 u[2]; v16bf v; };

constexpr int Bc = 4;        // batch
constexpr int Sc = 2048;     // sequence
constexpr int Dc = 1024;     // model dim
constexpr int Hc = 16;       // heads
constexpr int HDc = 64;      // head dim
constexpr int Mrows = Bc * Sc;        // 8192
constexpr float SCALEc = 0.125f;      // 1/sqrt(64)

__device__ __forceinline__ u16 f2bf(float f) {
  union { float f; unsigned u; } x; x.f = f;
  unsigned r = x.u + 0x7FFFu + ((x.u >> 16) & 1u);   // round-to-nearest-even
  return (u16)(r >> 16);
}

// Async DMA: global -> LDS, 16 bytes per active lane. LDS operand is the
// 32-bit LDS offset (= low 32 bits of the generic pointer to __shared__).
__device__ __forceinline__ void async_ld_b128(void* lds_ptr, const void* gptr) {
  unsigned lds_off = (unsigned)(unsigned long long)lds_ptr;
  asm volatile("global_load_async_to_lds_b128 %0, %1, off"
               :: "v"(lds_off), "v"(gptr) : "memory");
}
__device__ __forceinline__ void wait_async0() {
  asm volatile("s_wait_asynccnt 0" ::: "memory");
}

// ---------------------------------------------------------------------------
// fp32 -> bf16 elementwise
// ---------------------------------------------------------------------------
__global__ void cvt_f32_bf16(const float* __restrict__ src, u16* __restrict__ dst, int n) {
  int i = blockIdx.x * 256 + threadIdx.x;
  if (i < n) dst[i] = f2bf(src[i]);
}

// ---------------------------------------------------------------------------
// C[M,N] = A[M,K](bf16,row-major) @ Bm[N,K](bf16,row-major)^T + bias[N]
// Block: 256 threads = 8 waves (2 x 4). Block tile 64x256, wave tile 32x64.
// Async double-buffered LDS K-slices of 32. Output bf16 (storeBf16=1) or f32.
// ---------------------------------------------------------------------------
#define GBM 64
#define GBN 256
#define GBK 32
#define GPAD 8   // row stride (GBK+GPAD)*2B = 80B, multiple of 16B

__global__ void __launch_bounds__(256)
gemm_bf16(const u16* __restrict__ A, const u16* __restrict__ Bm,
          const float* __restrict__ bias, void* __restrict__ C,
          int M, int N, int K, int storeBf16)
{
  __shared__ u16 As[2][GBM][GBK + GPAD];
  __shared__ u16 Bs[2][GBN][GBK + GPAD];

  const int t = threadIdx.x;
  const int wave = t >> 5, lane = t & 31;
  const int wm = wave >> 2, wn = wave & 3;   // 2 x 4 wave grid
  const int hi = lane >> 4, lo = lane & 15;
  const int m0 = blockIdx.x * GBM;
  const int n0 = blockIdx.y * GBN;

  // issue one block tile's async loads into buffer `bf` for K-slice k0
  auto issue_tile = [&](int bf, int k0) {
    { // A tile: 64 rows x 32 cols -> 256 chunks of 8 bf16 (one per thread)
      int row = t >> 2, ch = (t & 3) * 8;
      async_ld_b128(&As[bf][row][ch], &A[(size_t)(m0 + row) * K + k0 + ch]);
    }
#pragma unroll
    for (int j = 0; j < 4; ++j) { // B tile: 256 x 32 -> 1024 chunks, 4 per thread
      int c = t + 256 * j;
      int row = c >> 2, ch = (c & 3) * 8;
      async_ld_b128(&Bs[bf][row][ch], &Bm[(size_t)(n0 + row) * K + k0 + ch]);
    }
  };

  v8f acc[2][4] = {};

  issue_tile(0, 0);
  wait_async0();
  __syncthreads();

  int buf = 0;
  for (int k0 = 0; k0 < K; k0 += GBK) {
    if (k0 + GBK < K) issue_tile(buf ^ 1, k0 + GBK);   // DMA overlaps WMMAs

    FragBF a[2], b[4];
#pragma unroll
    for (int i = 0; i < 2; ++i) {
      const u16* p = &As[buf][wm * 32 + i * 16 + lo][hi * 8];
      a[i].u[0] = *(const uint4*)p;
      a[i].u[1] = *(const uint4*)(p + 16);
    }
#pragma unroll
    for (int j = 0; j < 4; ++j) {
      const u16* p = &Bs[buf][wn * 64 + j * 16 + lo][hi * 8];
      b[j].u[0] = *(const uint4*)p;
      b[j].u[1] = *(const uint4*)(p + 16);
    }
#pragma unroll
    for (int i = 0; i < 2; ++i)
#pragma unroll
      for (int j = 0; j < 4; ++j)
        acc[i][j] = __builtin_amdgcn_wmma_f32_16x16x32_bf16(
            false, a[i].v, false, b[j].v, (short)0, acc[i][j], false, false);

    wait_async0();      // next buffer's DMA complete (this thread)
    __syncthreads();    // ... for all threads; also fences frag reads of `buf`
    buf ^= 1;
  }

  // Epilogue: C layout -> row = vgpr + 8*hi within 16-tile, col = lo
#pragma unroll
  for (int i = 0; i < 2; ++i) {
    int gmb = m0 + wm * 32 + i * 16 + 8 * hi;
#pragma unroll
    for (int j = 0; j < 4; ++j) {
      int gn = n0 + wn * 64 + j * 16 + lo;
      float bv = bias[gn];
#pragma unroll
      for (int v = 0; v < 8; ++v) {
        float val = acc[i][j][v] + bv;
        size_t idx = (size_t)(gmb + v) * N + gn;
        if (storeBf16) ((u16*)C)[idx] = f2bf(val);
        else           ((float*)C)[idx] = val;
      }
    }
  }
}

// ---------------------------------------------------------------------------
// Flash attention for one (b,h): 128 query rows per block (16 per wave),
// key blocks of 64 staged in LDS (K via async DMA, V transposed via regs).
// qkv is [8192, 3072] bf16: q at col h*64, k at 1024+h*64, v at 2048+h*64.
// attn_out bf16 [8192, 1024] (=[b,s,h*64+hd]).
// ---------------------------------------------------------------------------
__global__ void __launch_bounds__(256)
attn_flash(const u16* __restrict__ qkv, const float* __restrict__ bias,
           u16* __restrict__ attn_out)
{
  __shared__ u16 Ks[64][72];        // [key][hd],  144B rows
  __shared__ u16 Vs[64][72];        // [hd][key]  (transposed on fill)
  __shared__ u16 Ps[8][16][72];     // per-wave P tile [qrow][key]

  const int t = threadIdx.x;
  const int wave = t >> 5, lane = t & 31;
  const int hi = lane >> 4, lo = lane & 15;
  const int b = (int)blockIdx.y >> 4;
  const int h = (int)blockIdx.y & 15;
  const int qbase = (int)blockIdx.x * 128 + wave * 16;

  // Q fragments for this wave's 16 rows (K-dim = head dim 64 -> 2 frags)
  FragBF qf[2];
  {
    const u16* qrow = qkv + (size_t)(b * Sc + qbase + lo) * 3072 + h * HDc;
#pragma unroll
    for (int kk = 0; kk < 2; ++kk) {
      qf[kk].u[0] = *(const uint4*)&qrow[kk * 32 + hi * 8];
      qf[kk].u[1] = *(const uint4*)&qrow[kk * 32 + hi * 8 + 16];
    }
  }

  const float* brow0 = bias + ((size_t)h * Sc + (qbase + 8 * hi)) * Sc;

  float mrow[8], lrow[8], alpha[8];
  v8f O[4] = {};
#pragma unroll
  for (int i = 0; i < 8; ++i) { mrow[i] = -1e30f; lrow[i] = 0.0f; }

  for (int kb = 0; kb < Sc; kb += 64) {
    __syncthreads();
#pragma unroll
    for (int j = 0; j < 2; ++j) {   // K tile: 64x64 bf16 -> async DMA to LDS
      int c = t + 256 * j;
      int row = c >> 3, ch = (c & 7) * 8;
      async_ld_b128(&Ks[row][ch],
          &qkv[(size_t)(b * Sc + kb + row) * 3072 + 1024 + h * HDc + ch]);
    }
#pragma unroll
    for (int j = 0; j < 2; ++j) {   // V tile: load rows, scatter transposed
      int c = t + 256 * j;
      int row = c >> 3, ch = (c & 7) * 8;
      union { uint4 u; u16 s[8]; } tmp;
      tmp.u = *(const uint4*)&qkv[(size_t)(b * Sc + kb + row) * 3072 + 2048 + h * HDc + ch];
#pragma unroll
      for (int e = 0; e < 8; ++e) Vs[ch + e][row] = tmp.s[e];
    }
    // prefetch next key-block's bias rows (dominant HBM stream) while we compute
    if (kb + 64 < Sc) {
      const float* pf = brow0 + (kb + 64) + lo * 4;   // lanes 16B apart -> 256B/row
#pragma unroll
      for (int i = 0; i < 8; ++i)
        __builtin_prefetch(pf + (size_t)i * Sc, 0, 3);
    }
    wait_async0();
    __syncthreads();

    // S = Q @ K^T  (4 accumulators of 16 queries x 16 keys)
    v8f Sv[4] = {};
#pragma unroll
    for (int kk = 0; kk < 2; ++kk) {
#pragma unroll
      for (int tn = 0; tn < 4; ++tn) {
        FragBF kf;
        const u16* p = &Ks[tn * 16 + lo][kk * 32 + hi * 8];
        kf.u[0] = *(const uint4*)p;
        kf.u[1] = *(const uint4*)(p + 16);
        Sv[tn] = __builtin_amdgcn_wmma_f32_16x16x32_bf16(
            false, qf[kk].v, false, kf.v, (short)0, Sv[tn], false, false);
      }
    }

    // scale + bias, then online softmax
    float sc[4][8];
#pragma unroll
    for (int tn = 0; tn < 4; ++tn) {
      const float* bp = brow0 + kb + tn * 16 + lo;
#pragma unroll
      for (int i = 0; i < 8; ++i)
        sc[tn][i] = Sv[tn][i] * SCALEc + bp[(size_t)i * Sc];
    }
#pragma unroll
    for (int i = 0; i < 8; ++i) {
      float m = sc[0][i];
#pragma unroll
      for (int tn = 1; tn < 4; ++tn) m = fmaxf(m, sc[tn][i]);
      for (int d = 1; d < 16; d <<= 1) m = fmaxf(m, __shfl_xor(m, d, 16));
      float mnew = fmaxf(mrow[i], m);
      alpha[i] = __expf(mrow[i] - mnew);
      mrow[i] = mnew;
    }
    float rsum[8];
#pragma unroll
    for (int i = 0; i < 8; ++i) rsum[i] = 0.0f;
#pragma unroll
    for (int tn = 0; tn < 4; ++tn)
#pragma unroll
      for (int i = 0; i < 8; ++i) {
        float p = __expf(sc[tn][i] - mrow[i]);
        sc[tn][i] = p;
        rsum[i] += p;
      }
#pragma unroll
    for (int i = 0; i < 8; ++i) {
      float r = rsum[i];
      for (int d = 1; d < 16; d <<= 1) r += __shfl_xor(r, d, 16);
      lrow[i] = lrow[i] * alpha[i] + r;
#pragma unroll
      for (int tn = 0; tn < 4; ++tn) O[tn][i] *= alpha[i];
    }

    // P (C-layout) -> LDS -> A-fragment layout, bf16
#pragma unroll
    for (int tn = 0; tn < 4; ++tn)
#pragma unroll
      for (int i = 0; i < 8; ++i)
        Ps[wave][i + 8 * hi][tn * 16 + lo] = f2bf(sc[tn][i]);
    __syncthreads();

    // O += P @ V   (K-dim = 64 keys -> 2 frags)
#pragma unroll
    for (int kk = 0; kk < 2; ++kk) {
      FragBF pf;
      const u16* pp = &Ps[wave][lo][kk * 32 + hi * 8];
      pf.u[0] = *(const uint4*)pp;
      pf.u[1] = *(const uint4*)(pp + 16);
#pragma unroll
      for (int tn = 0; tn < 4; ++tn) {
        FragBF vf;
        const u16* vp = &Vs[tn * 16 + lo][kk * 32 + hi * 8];
        vf.u[0] = *(const uint4*)vp;
        vf.u[1] = *(const uint4*)(vp + 16);
        O[tn] = __builtin_amdgcn_wmma_f32_16x16x32_bf16(
            false, pf.v, false, vf.v, (short)0, O[tn], false, false);
      }
    }
  }

  // normalize and store bf16 at [b, q, h*64 + hd]
#pragma unroll
  for (int i = 0; i < 8; ++i) {
    float inv = 1.0f / lrow[i];
    int q = qbase + i + 8 * hi;
    u16* op = attn_out + (size_t)(b * Sc + q) * Dc + h * HDc;
#pragma unroll
    for (int tn = 0; tn < 4; ++tn)
      op[tn * 16 + lo] = f2bf(O[tn][i] * inv);
  }
}

// ---------------------------------------------------------------------------
extern "C" void kernel_launch(void* const* d_in, const int* in_sizes, int n_in,
                              void* d_out, int out_size, void* d_ws, size_t ws_size,
                              hipStream_t stream) {
  (void)in_sizes; (void)n_in; (void)out_size; (void)ws_size;
  const float* x      = (const float*)d_in[0];
  const float* abias  = (const float*)d_in[1];
  const float* qkv_w  = (const float*)d_in[2];
  const float* qkv_b  = (const float*)d_in[3];
  const float* out_w  = (const float*)d_in[4];
  const float* out_b  = (const float*)d_in[5];

  // workspace layout (bf16 = u16), ~92 MB total
  u16* xb   = (u16*)d_ws;                              //  8192*1024
  u16* wqkv = xb   + (size_t)Mrows * Dc;               //  3072*1024
  u16* wout = wqkv + (size_t)3 * Dc * Dc;              //  1024*1024
  u16* qkv  = wout + (size_t)Dc * Dc;                  //  8192*3072
  u16* attn = qkv  + (size_t)Mrows * 3 * Dc;           //  8192*1024

  cvt_f32_bf16<<<(Mrows * Dc) / 256, 256, 0, stream>>>(x, xb, Mrows * Dc);
  cvt_f32_bf16<<<(3 * Dc * Dc) / 256, 256, 0, stream>>>(qkv_w, wqkv, 3 * Dc * Dc);
  cvt_f32_bf16<<<(Dc * Dc) / 256, 256, 0, stream>>>(out_w, wout, Dc * Dc);

  // qkv projection: [8192,3072] bf16
  gemm_bf16<<<dim3(Mrows / GBM, (3 * Dc) / GBN), 256, 0, stream>>>(
      xb, wqkv, qkv_b, qkv, Mrows, 3 * Dc, Dc, 1);

  // attention: grid (S/128, B*H)
  attn_flash<<<dim3(Sc / 128, Bc * Hc), 256, 0, stream>>>(qkv, abias, attn);

  // output projection: fp32 out
  gemm_bf16<<<dim3(Mrows / GBM, Dc / GBN), 256, 0, stream>>>(
      attn, wout, out_b, (float*)d_out, Mrows, Dc, Dc, 0);
}
